// AnalyticHashLinear_64192581206209
// MI455X (gfx1250) — matrix operand order repaired and checked by reading, not compiled, not run on hardware
//
#include <hip/hip_runtime.h>

// ---------------------------------------------------------------------------
// AnalyticHashLinear: y = x @ W^T + bias, W generated from a 64K-entry
// codebook via affine hashes.  8192x4096x4096 GEMM -> compute-bound on
// MI455X (AI ~1000 FLOP/B vs machine balance ~100 FLOP/B) => single-pass
// f16 WMMA (v_wmma_f32_16x16x32_f16) with f32 accumulation.
// Phase 1 materializes W (f16, 32MB) and x (f16, 64MB) into d_ws; phase 2 is
// a tiled WMMA GEMM (block 128x256, BK=64, 8 wave32 waves, 64x64 per-wave
// tile) where BOTH tiles stream global->LDS via CDNA5 async DMA
// (global_load_async_to_lds_b128, ASYNCcnt) into double-buffered LDS with
// one barrier per K-step.
// ---------------------------------------------------------------------------

typedef __attribute__((ext_vector_type(4)))  int      v4i;

#if defined(__AMDGCN__) && \
    __has_builtin(__builtin_amdgcn_global_load_async_to_lds_b128) && \
    __has_builtin(__builtin_amdgcn_s_wait_asynccnt)
#define USE_ASYNC_LDS 1
typedef __attribute__((address_space(1))) v4i g_v4i;   // global int4
typedef __attribute__((address_space(3))) v4i l_v4i;   // LDS int4
#else
#define USE_ASYNC_LDS 0
#endif

typedef __attribute__((ext_vector_type(16))) _Float16 v16h;
typedef __attribute__((ext_vector_type(8)))  _Float16 v8h;
typedef __attribute__((ext_vector_type(4)))  _Float16 v4h;
typedef __attribute__((ext_vector_type(8)))  float    v8f;

#define IN_DIM   4096
#define OUT_DIM  4096
#define BATCH    8192
#define CB_MASK  0xFFFF
#define HASH_A   10007
#define HASH_B   20011
#define HASH_C   40009
#define SHASH_A  4099
#define SHASH_B  6151
#define SHASH_C  14887
#define LAYER_ID 3

#define BM    128
#define BN    256
#define BK    64
#define LDT   72        // padded LDS row stride (elements) to stagger banks
#define NSTEP (IN_DIM / BK)

// ---------------------------------------------------------------------------
// Phase 1a: materialize W[o][i] = sign * codebook[hash] as f16, row-major.
// ---------------------------------------------------------------------------
__global__ __launch_bounds__(256)
void ahl_build_w(const float* __restrict__ codebook, _Float16* __restrict__ w) {
    const int e = (blockIdx.x * 256 + threadIdx.x) * 8;   // element index
    const int o = e >> 12;                                // / IN_DIM
    const int i = e & (IN_DIM - 1);
    const int hbase = o * HASH_A  + i * HASH_B  + LAYER_ID * HASH_C;
    const int sbase = o * SHASH_A + i * SHASH_B + LAYER_ID * SHASH_C;
    v8h hv;
#pragma unroll
    for (int t = 0; t < 8; ++t) {
        const float c   = codebook[(hbase + t * HASH_B) & CB_MASK];
        const int   bit = (sbase + t * SHASH_B) & 1;      // bit=1 -> +1, bit=0 -> -1
        hv[t] = (_Float16)(bit ? c : -c);
    }
    *reinterpret_cast<v8h*>(w + e) = hv;
}

// ---------------------------------------------------------------------------
// Phase 1b: convert x (f32) -> xh (f16), 8 elements / thread.
// ---------------------------------------------------------------------------
__global__ __launch_bounds__(256)
void ahl_build_xh(const float* __restrict__ x, _Float16* __restrict__ xh) {
    const int e = (blockIdx.x * 256 + threadIdx.x) * 8;
    const float4 a = *reinterpret_cast<const float4*>(x + e);
    const float4 b = *reinterpret_cast<const float4*>(x + e + 4);
    v8h hv;
    hv[0] = (_Float16)a.x; hv[1] = (_Float16)a.y;
    hv[2] = (_Float16)a.z; hv[3] = (_Float16)a.w;
    hv[4] = (_Float16)b.x; hv[5] = (_Float16)b.y;
    hv[6] = (_Float16)b.z; hv[7] = (_Float16)b.w;
    *reinterpret_cast<v8h*>(xh + e) = hv;
}

// ---------------------------------------------------------------------------
// Shared compute + epilogue helpers (64x64 per-wave tile, 4x4 WMMA grid).
// ---------------------------------------------------------------------------
#define AHL_COMPUTE_TILE(sAc, sBc)                                             \
    _Pragma("unroll")                                                          \
    for (int kc = 0; kc < BK; kc += 32) {                                      \
        v16h a[4];                                                             \
        _Pragma("unroll")                                                      \
        for (int mi = 0; mi < 4; ++mi) {                                       \
            const _Float16* p = &(sAc)[(wm + mi * 16 + lane15) * LDT + kc + aKb]; \
            const v8h lo = *reinterpret_cast<const v8h*>(p);                   \
            const v8h hi = *reinterpret_cast<const v8h*>(p + 16);              \
            a[mi] = __builtin_shufflevector(lo, hi,                            \
                    0, 1, 2, 3, 4, 5, 6, 7, 8, 9, 10, 11, 12, 13, 14, 15);    \
        }                                                                      \
        _Pragma("unroll")                                                      \
        for (int ni = 0; ni < 4; ++ni) {                                       \
            const _Float16* p = &(sBc)[(wn + ni * 16 + lane15) * LDT + kc + bKb]; \
            const v8h lo = *reinterpret_cast<const v8h*>(p);                   \
            const v8h hi = *reinterpret_cast<const v8h*>(p + 16);              \
            const v16h b = __builtin_shufflevector(lo, hi,                     \
                    0, 1, 2, 3, 4, 5, 6, 7, 8, 9, 10, 11, 12, 13, 14, 15);    \
            _Pragma("unroll")                                                  \
            for (int mi = 0; mi < 4; ++mi) {                                   \
                acc[mi][ni] = __builtin_amdgcn_wmma_f32_16x16x32_f16(          \
                    false, a[mi], false, b, (short)0, acc[mi][ni],             \
                    false, false);                                             \
            }                                                                  \
        }                                                                      \
    }

#define AHL_EPILOGUE()                                                         \
    {                                                                          \
        const int mOff = (lane >> 4) ? 8 : 0;                                  \
        _Pragma("unroll")                                                      \
        for (int mi = 0; mi < 4; ++mi) {                                       \
            _Pragma("unroll")                                                  \
            for (int ni = 0; ni < 4; ++ni) {                                   \
                const int   n  = n0 + wn + ni * 16 + lane15;                   \
                const float bv = bias[n];                                      \
                const int   mb = m0 + wm + mi * 16 + mOff;                     \
                _Pragma("unroll")                                              \
                for (int r = 0; r < 8; ++r)                                    \
                    out[(size_t)(mb + r) * OUT_DIM + n] = acc[mi][ni][r] + bv; \
            }                                                                  \
        }                                                                      \
    }

// ---------------------------------------------------------------------------
// Phase 2 (primary): both x (f16) and W (f16) stream via async DMA into
// double-buffered LDS; one barrier per K-step; DMA overlaps 32 WMMAs.
// ---------------------------------------------------------------------------
__global__ __launch_bounds__(256)
void ahl_gemm_xh(const _Float16* __restrict__ xh, const _Float16* __restrict__ w,
                 const float* __restrict__ bias, float* __restrict__ out) {
    __shared__ _Float16 sA[2][BM * LDT];     // x tile, 36 KB
    __shared__ _Float16 sB[2][BN * LDT];     // W tile, 72 KB

    const int tid    = threadIdx.x;
    const int lane   = tid & 31;
    const int wave   = tid >> 5;
    const int m0     = blockIdx.y * BM;
    const int n0     = blockIdx.x * BN;
    const int wm     = (wave & 1) * 64;
    const int wn     = (wave >> 1) * 64;
    const int lane15 = lane & 15;
    const int aKb    = (lane >> 4) ? 8  : 0;
    const int bKb    = (lane >> 4) ? 8  : 0;

    v8f acc[4][4] = {};

#if USE_ASYNC_LDS
    // ---- prologue: async DMA tile 0 into buffer 0
#pragma unroll
    for (int it = 0; it < 4; ++it) {        // x: 1024 16B-chunks
        const int f = tid + it * 256;
        __builtin_amdgcn_global_load_async_to_lds_b128(
            (g_v4i*)(xh + (size_t)(m0 + (f >> 3)) * IN_DIM + (f & 7) * 8),
            (l_v4i*)&sA[0][(f >> 3) * LDT + (f & 7) * 8], 0, 0);
    }
#pragma unroll
    for (int it = 0; it < 8; ++it) {        // W: 2048 16B-chunks
        const int f = tid + it * 256;
        __builtin_amdgcn_global_load_async_to_lds_b128(
            (g_v4i*)(w + (size_t)(n0 + (f >> 3)) * IN_DIM + (f & 7) * 8),
            (l_v4i*)&sB[0][(f >> 3) * LDT + (f & 7) * 8], 0, 0);
    }
    int cur = 0;
    for (int s = 0; s < NSTEP; ++s) {
        __builtin_amdgcn_s_wait_asynccnt(0);  // this wave's tile-s DMAs landed
        __syncthreads();                      // tile s visible; buf cur^1 free
        if (s + 1 < NSTEP) {                  // DMA tile s+1, overlaps WMMAs
            const int kn = (s + 1) * BK;
#pragma unroll
            for (int it = 0; it < 4; ++it) {
                const int f = tid + it * 256;
                __builtin_amdgcn_global_load_async_to_lds_b128(
                    (g_v4i*)(xh + (size_t)(m0 + (f >> 3)) * IN_DIM + kn + (f & 7) * 8),
                    (l_v4i*)&sA[cur ^ 1][(f >> 3) * LDT + (f & 7) * 8], 0, 0);
            }
#pragma unroll
            for (int it = 0; it < 8; ++it) {
                const int f = tid + it * 256;
                __builtin_amdgcn_global_load_async_to_lds_b128(
                    (g_v4i*)(w + (size_t)(n0 + (f >> 3)) * IN_DIM + kn + (f & 7) * 8),
                    (l_v4i*)&sB[cur ^ 1][(f >> 3) * LDT + (f & 7) * 8], 0, 0);
            }
        }
        const _Float16* sAc = sA[cur];
        const _Float16* sBc = sB[cur];
        AHL_COMPUTE_TILE(sAc, sBc)
        cur ^= 1;
    }
#else
    // ---- synchronous fallback (toolchain without async-LDS builtins)
    int cur = 0;
    for (int s = 0; s < NSTEP; ++s) {
        const int k0 = s * BK;
        __syncthreads();
#pragma unroll
        for (int it = 0; it < 4; ++it) {
            const int f = tid + it * 256;
            *reinterpret_cast<v8h*>(&sA[cur][(f >> 3) * LDT + (f & 7) * 8]) =
                *reinterpret_cast<const v8h*>(
                    xh + (size_t)(m0 + (f >> 3)) * IN_DIM + k0 + (f & 7) * 8);
        }
#pragma unroll
        for (int it = 0; it < 8; ++it) {
            const int f = tid + it * 256;
            *reinterpret_cast<v8h*>(&sB[cur][(f >> 3) * LDT + (f & 7) * 8]) =
                *reinterpret_cast<const v8h*>(
                    w + (size_t)(n0 + (f >> 3)) * IN_DIM + k0 + (f & 7) * 8);
        }
        __syncthreads();
        const _Float16* sAc = sA[cur];
        const _Float16* sBc = sB[cur];
        AHL_COMPUTE_TILE(sAc, sBc)
        cur ^= 1;
    }
#endif

    AHL_EPILOGUE()
}

// ---------------------------------------------------------------------------
// Phase 2 (fallbacks): x staged from f32 via registers; W from ws (async,
// double-buffered) or generated in-kernel (GEN=true).
// ---------------------------------------------------------------------------
template <bool GEN>
__global__ __launch_bounds__(256)
void ahl_gemm(const float* __restrict__ x, const _Float16* __restrict__ w,
              const float* __restrict__ codebook, const float* __restrict__ bias,
              float* __restrict__ out) {
    __shared__ _Float16 sA[BM * LDT];        // x tile (f16), 18 KB
    __shared__ _Float16 sB[2][BN * LDT];     // W tile (f16), double-buffered 72 KB

    const int tid    = threadIdx.x;
    const int lane   = tid & 31;
    const int wave   = tid >> 5;
    const int m0     = blockIdx.y * BM;
    const int n0     = blockIdx.x * BN;
    const int wm     = (wave & 1) * 64;
    const int wn     = (wave >> 1) * 64;
    const int lane15 = lane & 15;
    const int aKb    = (lane >> 4) ? 8  : 0;
    const int bKb    = (lane >> 4) ? 8  : 0;

    v8f acc[4][4] = {};

    float4 xs[8];                             // pipelined x tile (regs)
#if !USE_ASYNC_LDS
    v8h wv[8];                                // fallback: pipelined W tile (regs)
#endif

#pragma unroll
    for (int it = 0; it < 8; ++it) {
        const int f = tid + it * 256;
        xs[it] = *reinterpret_cast<const float4*>(
            x + (size_t)(m0 + (f >> 4)) * IN_DIM + (f & 15) * 4);
    }
    if constexpr (!GEN) {
#if USE_ASYNC_LDS
#pragma unroll
        for (int it = 0; it < 8; ++it) {
            const int f = tid + it * 256;
            __builtin_amdgcn_global_load_async_to_lds_b128(
                (g_v4i*)(w + (size_t)(n0 + (f >> 3)) * IN_DIM + (f & 7) * 8),
                (l_v4i*)&sB[0][(f >> 3) * LDT + (f & 7) * 8], 0, 0);
        }
#else
#pragma unroll
        for (int it = 0; it < 8; ++it) {
            const int f = tid + it * 256;
            wv[it] = *reinterpret_cast<const v8h*>(
                w + (size_t)(n0 + (f >> 3)) * IN_DIM + (f & 7) * 8);
        }
#endif
    }

    int cur = 0;
    for (int s = 0; s < NSTEP; ++s) {
        const int k0 = s * BK;
        __syncthreads();
#pragma unroll
        for (int it = 0; it < 8; ++it) {
            const int f   = tid + it * 256;
            const int row = f >> 4;
            const int c4  = (f & 15) * 4;
            v4h h;
            h[0] = (_Float16)xs[it].x; h[1] = (_Float16)xs[it].y;
            h[2] = (_Float16)xs[it].z; h[3] = (_Float16)xs[it].w;
            *reinterpret_cast<v4h*>(&sA[row * LDT + c4]) = h;
        }
        if constexpr (!GEN) {
#if !USE_ASYNC_LDS
#pragma unroll
            for (int it = 0; it < 8; ++it) {
                const int f = tid + it * 256;
                *reinterpret_cast<v8h*>(&sB[cur][(f >> 3) * LDT + (f & 7) * 8]) = wv[it];
            }
#endif
        } else {
#pragma unroll
            for (int it = 0; it < 8; ++it) {
                const int f   = tid + it * 256;
                const int row = f >> 3;
                const int c8  = (f & 7) * 8;
                const int o   = n0 + row;
                const int ib  = k0 + c8;
                const int hb  = o * HASH_A  + ib * HASH_B  + LAYER_ID * HASH_C;
                const int sb  = o * SHASH_A + ib * SHASH_B + LAYER_ID * SHASH_C;
                v8h hv;
#pragma unroll
                for (int t = 0; t < 8; ++t) {
                    const float c   = codebook[(hb + t * HASH_B) & CB_MASK];
                    const int   bit = (sb + t * SHASH_B) & 1;
                    hv[t] = (_Float16)(bit ? c : -c);
                }
                *reinterpret_cast<v8h*>(&sB[cur][row * LDT + c8]) = hv;
            }
        }
        if (s + 1 < NSTEP) {
            const int kn = k0 + BK;
#pragma unroll
            for (int it = 0; it < 8; ++it) {
                const int f = tid + it * 256;
                xs[it] = *reinterpret_cast<const float4*>(
                    x + (size_t)(m0 + (f >> 4)) * IN_DIM + kn + (f & 15) * 4);
            }
        }
#if USE_ASYNC_LDS
        if constexpr (!GEN) __builtin_amdgcn_s_wait_asynccnt(0);
#endif
        __syncthreads();

        if constexpr (!GEN) {
            if (s + 1 < NSTEP) {
                const int kn = k0 + BK;
#if USE_ASYNC_LDS
#pragma unroll
                for (int it = 0; it < 8; ++it) {
                    const int f = tid + it * 256;
                    __builtin_amdgcn_global_load_async_to_lds_b128(
                        (g_v4i*)(w + (size_t)(n0 + (f >> 3)) * IN_DIM + kn + (f & 7) * 8),
                        (l_v4i*)&sB[cur ^ 1][(f >> 3) * LDT + (f & 7) * 8], 0, 0);
                }
#else
#pragma unroll
                for (int it = 0; it < 8; ++it) {
                    const int f = tid + it * 256;
                    wv[it] = *reinterpret_cast<const v8h*>(
                        w + (size_t)(n0 + (f >> 3)) * IN_DIM + kn + (f & 7) * 8);
                }
#endif
            }
        }

        const _Float16* sAc = sA;
        const _Float16* sBc = sB[cur];
        AHL_COMPUTE_TILE(sAc, sBc)
        cur ^= 1;
    }

    AHL_EPILOGUE()
}

// ---------------------------------------------------------------------------
extern "C" void kernel_launch(void* const* d_in, const int* in_sizes, int n_in,
                              void* d_out, int out_size, void* d_ws, size_t ws_size,
                              hipStream_t stream) {
    const float* x        = (const float*)d_in[0];
    const float* codebook = (const float*)d_in[1];
    const float* bias     = (const float*)d_in[2];
    float*       out      = (float*)d_out;

    const size_t wbytes = (size_t)OUT_DIM * IN_DIM * sizeof(_Float16);  // 32 MB
    const size_t xbytes = (size_t)BATCH   * IN_DIM * sizeof(_Float16);  // 64 MB
    dim3 grid(OUT_DIM / BN, BATCH / BM);   // 16 x 64 workgroups

    if (ws_size >= wbytes + xbytes) {
        _Float16* w  = (_Float16*)d_ws;
        _Float16* xh = (_Float16*)((char*)d_ws + wbytes);
        ahl_build_w<<<(OUT_DIM / 8) * (IN_DIM / 256), 256, 0, stream>>>(codebook, w);
        ahl_build_xh<<<(BATCH / 8) * (IN_DIM / 256), 256, 0, stream>>>(x, xh);
        ahl_gemm_xh<<<grid, 256, 0, stream>>>(xh, w, bias, out);
    } else if (ws_size >= wbytes) {
        _Float16* w = (_Float16*)d_ws;
        ahl_build_w<<<(OUT_DIM / 8) * (IN_DIM / 256), 256, 0, stream>>>(codebook, w);
        ahl_gemm<false><<<grid, 256, 0, stream>>>(x, w, codebook, bias, out);
    } else {
        ahl_gemm<true><<<grid, 256, 0, stream>>>(x, nullptr, codebook, bias, out);
    }
}
